// GeneralConv_936302871059
// MI455X (gfx1250) — matrix operand order repaired and checked by reading, not compiled, not run on hardware
//
#include <hip/hip_runtime.h>
#include <hip/hip_bf16.h>

#define N_NODES 50000
#define N_EDGES 800000
#define C_IN    128
#define C_OUT   128
#define HEADS   4
#define K_DIM   128
// Packed GEMM output columns: [0,128)=P_src  [128,256)=P_dst  [256,384)=Root
// [384,388)=a_src  [388,392)=a_dst  [392,416)=zero pad  (26 tiles of 16)
#define NC      416

typedef __attribute__((ext_vector_type(16))) __bf16 v16bf;
typedef __attribute__((ext_vector_type(8)))  __bf16 v8bf;   // 16 bytes -> b128 load
typedef __attribute__((ext_vector_type(8)))  float  v8f;

union BF16x16 { v16bf v; v8bf h[2]; };

// ---------------- init: zero out, init segmax/denom ----------------
__global__ void init_kernel(float* __restrict__ out, float* __restrict__ segmax,
                            float* __restrict__ denom) {
    int i = blockIdx.x * blockDim.x + threadIdx.x;
    if (i < N_NODES * C_OUT) out[i] = 0.0f;
    if (i < N_NODES * HEADS) { segmax[i] = -1.0e30f; denom[i] = 0.0f; }
}

// ---------------- fp32 -> bf16 convert of x ----------------
__global__ void convert_x(const float* __restrict__ x, __bf16* __restrict__ xb) {
    int i = blockIdx.x * blockDim.x + threadIdx.x;
    if (i < N_NODES * C_IN) xb[i] = (__bf16)x[i];
}

// ---------------- pack all weights, transposed, bf16: Bt[col][k] ----------------
__global__ void pack_weights(const float* __restrict__ Wm, const float* __restrict__ Wa,
                             const float* __restrict__ Wr, __bf16* __restrict__ Bt) {
    int i = blockIdx.x * blockDim.x + threadIdx.x;
    if (i >= NC * K_DIM) return;
    int r = i / K_DIM;   // output column 0..415
    int k = i % K_DIM;   // input channel
    float v = 0.0f;
    if      (r < 128) v = Wm[k * C_OUT + r];                     // W_msg top (src)
    else if (r < 256) v = Wm[(128 + k) * C_OUT + (r - 128)];     // W_msg bottom (dst)
    else if (r < 384) v = Wr[k * C_OUT + (r - 256)];             // W_root
    else if (r < 388) v = Wa[k * HEADS + (r - 384)];             // W_att top
    else if (r < 392) v = Wa[(128 + k) * HEADS + (r - 388)];     // W_att bottom
    Bt[r * K_DIM + k] = (__bf16)v;
}

// ---------------- WMMA GEMM: P[N,416] = xb[N,128] @ Bt^T ----------------
// 8 waves/block, one 16x16 M-tile x TWO 16-wide N-tiles per wave (A reg reuse).
// K=128 in 4 steps of 32 -> 8 v_wmma per wave.
__global__ void __launch_bounds__(256) gemm_wmma(const __bf16* __restrict__ xb,
                                                 const __bf16* __restrict__ Bt,
                                                 float* __restrict__ P) {
    const int lane = threadIdx.x & 31;
    const int wave = threadIdx.x >> 5;
    const int mt = blockIdx.x * 8 + wave;
    if (mt >= N_NODES / 16) return;            // wave-uniform; EXEC stays all-ones
    const int nt0 = blockIdx.y * 2;            // two consecutive N tiles

    // A, contiguous k order used here (16x32): lanes 0-15 row M=lane, K=0..15;
    //                          lanes 16-31 row M=lane-16, K=16..31
    const int arow = mt * 16 + (lane & 15);
    const int ak   = (lane >> 4) ? 16 : 0;
    // B, same contiguous k order as A (32x16): lane n (0-15) holds col n, K=0..15; lanes 16-31 K=16..31
    const int bcol0 = nt0 * 16 + (lane & 15);
    const int bk    = (lane >> 4) ? 16 : 0;

    v8f acc0 = {}, acc1 = {};
    #pragma unroll
    for (int kb = 0; kb < K_DIM; kb += 32) {
        BF16x16 a, b0, b1;
        a.h[0]  = *(const v8bf*)(xb + (size_t)arow * K_DIM + kb + ak);
        a.h[1]  = *(const v8bf*)(xb + (size_t)arow * K_DIM + kb + ak + 8);
        b0.h[0] = *(const v8bf*)(Bt + (size_t)bcol0 * K_DIM + kb + bk);
        b0.h[1] = *(const v8bf*)(Bt + (size_t)bcol0 * K_DIM + kb + bk + 8);
        b1.h[0] = *(const v8bf*)(Bt + (size_t)(bcol0 + 16) * K_DIM + kb + bk);
        b1.h[1] = *(const v8bf*)(Bt + (size_t)(bcol0 + 16) * K_DIM + kb + bk + 8);
        acc0 = __builtin_amdgcn_wmma_f32_16x16x32_bf16(false, a.v, false, b0.v,
                                                       (short)0, acc0, false, false);
        acc1 = __builtin_amdgcn_wmma_f32_16x16x32_bf16(false, a.v, false, b1.v,
                                                       (short)0, acc1, false, false);
    }
    // C layout: VGPR j -> M = j (lanes 0-15) / M = 8+j (lanes 16-31), N = lane&15
    const int row0 = mt * 16 + ((lane >> 4) ? 8 : 0);
    const int col0 = nt0 * 16 + (lane & 15);
    #pragma unroll
    for (int j = 0; j < 8; ++j) {
        P[(size_t)(row0 + j) * NC + col0]      = acc0[j];
        P[(size_t)(row0 + j) * NC + col0 + 16] = acc1[j];
    }
}

// ---------------- pass 1: per-head segment max via float atomicMax ----------------
__global__ void edge_score_max(const int* __restrict__ ei, const float* __restrict__ P,
                               float* __restrict__ segmax) {
    int e = blockIdx.x * blockDim.x + threadIdx.x;
    if (e >= N_EDGES) return;
    int r = ei[e];
    int c = ei[N_EDGES + e];
    const float* Pr = P + (size_t)r * NC + 384;
    const float* Pc = P + (size_t)c * NC + 388;
    #pragma unroll
    for (int h = 0; h < HEADS; ++h)
        atomicMax(segmax + c * HEADS + h, Pr[h] + Pc[h]);
}

// ---------------- pass 2: denom = segment_sum(exp(s - max)) ----------------
__global__ void edge_exp_sum(const int* __restrict__ ei, const float* __restrict__ P,
                             const float* __restrict__ segmax, float* __restrict__ denom) {
    int e = blockIdx.x * blockDim.x + threadIdx.x;
    if (e >= N_EDGES) return;
    int r = ei[e];
    int c = ei[N_EDGES + e];
    const float* Pr = P + (size_t)r * NC + 384;
    const float* Pc = P + (size_t)c * NC + 388;
    #pragma unroll
    for (int h = 0; h < HEADS; ++h) {
        float ex = __expf(Pr[h] + Pc[h] - segmax[c * HEADS + h]);
        atomicAdd(denom + c * HEADS + h, ex);
    }
}

// ---------------- pass 3: weighted message scatter (1 wave per edge) ----------------
__global__ void __launch_bounds__(256) edge_scatter(const int* __restrict__ ei,
                                                    const float* __restrict__ P,
                                                    const float* __restrict__ segmax,
                                                    const float* __restrict__ denom,
                                                    const float* __restrict__ bmsg,
                                                    float* __restrict__ out) {
    const int lane = threadIdx.x & 31;
    const int wave = threadIdx.x >> 5;
    const int e = blockIdx.x * 8 + wave;
    if (e >= N_EDGES) return;
    const int r = ei[e];
    const int c = ei[N_EDGES + e];
    const float* Pr = P + (size_t)r * NC;
    const float* Pc = P + (size_t)c * NC;
    float w[HEADS];
    #pragma unroll
    for (int h = 0; h < HEADS; ++h) {
        float ex = __expf(Pr[384 + h] + Pc[388 + h] - segmax[c * HEADS + h]);
        w[h] = ex / denom[c * HEADS + h];
    }
    #pragma unroll
    for (int h = 0; h < HEADS; ++h) {
        int ch = h * 32 + lane;                   // head h owns channels [32h, 32h+32)
        float m = Pr[ch] + Pc[128 + ch] + bmsg[ch];
        atomicAdd(out + (size_t)c * C_OUT + ch, m * w[h]);
    }
}

// ---------------- final: add root transform + bias ----------------
__global__ void add_root(const float* __restrict__ P, const float* __restrict__ broot,
                         float* __restrict__ out) {
    int i = blockIdx.x * blockDim.x + threadIdx.x;
    if (i >= N_NODES * C_OUT) return;
    int n = i >> 7, ch = i & 127;
    out[i] += P[(size_t)n * NC + 256 + ch] + broot[ch];
}

extern "C" void kernel_launch(void* const* d_in, const int* in_sizes, int n_in,
                              void* d_out, int out_size, void* d_ws, size_t ws_size,
                              hipStream_t stream) {
    const float* x     = (const float*)d_in[0];
    const int*   ei    = (const int*)  d_in[1];
    const float* Wmsg  = (const float*)d_in[2];
    const float* bmsg  = (const float*)d_in[3];
    const float* Watt  = (const float*)d_in[4];
    const float* Wroot = (const float*)d_in[5];
    const float* broot = (const float*)d_in[6];
    float* out = (float*)d_out;

    // workspace layout
    char* ws = (char*)d_ws;
    __bf16* xb     = (__bf16*)ws;                                    // 12,800,000 B
    __bf16* Bt     = (__bf16*)(ws + (size_t)N_NODES * C_IN * 2);     //    106,496 B
    float*  P      = (float*)(ws + (size_t)N_NODES * C_IN * 2 + (size_t)NC * K_DIM * 2);
    float*  segmax = (float*)((char*)P + (size_t)N_NODES * NC * 4);  //    800,000 B
    float*  denom  = (float*)((char*)segmax + (size_t)N_NODES * HEADS * 4);

    const int T = 256;
    init_kernel<<<(N_NODES * C_OUT + T - 1) / T, T, 0, stream>>>(out, segmax, denom);
    convert_x<<<(N_NODES * C_IN + T - 1) / T, T, 0, stream>>>(x, xb);
    pack_weights<<<(NC * K_DIM + T - 1) / T, T, 0, stream>>>(Wmsg, Watt, Wroot, Bt);

    dim3 ggrid((N_NODES / 16 + 7) / 8, NC / 32);                     // (391, 13)
    gemm_wmma<<<ggrid, T, 0, stream>>>(xb, Bt, P);

    edge_score_max<<<(N_EDGES + T - 1) / T, T, 0, stream>>>(ei, P, segmax);
    edge_exp_sum<<<(N_EDGES + T - 1) / T, T, 0, stream>>>(ei, P, segmax, denom);
    edge_scatter<<<(N_EDGES + 7) / 8, T, 0, stream>>>(ei, P, segmax, denom, bmsg, out);
    add_root<<<(N_NODES * C_OUT + T - 1) / T, T, 0, stream>>>(P, broot, out);
}
